// SelfNonParametric_Distil_70531952935520
// MI455X (gfx1250) — compile-verified, matrix operand order checked by reading, not run on hardware
//
#include <hip/hip_runtime.h>
#include <hip/hip_bf16.h>
#include <math.h>

typedef __attribute__((ext_vector_type(16))) __bf16 v16bf;
typedef __attribute__((ext_vector_type(8)))  __bf16 v8bf;
typedef __attribute__((ext_vector_type(8)))  float  v8f;

#define DIM      512
#define NROWS    8192
#define NLB      2048
#define NC       1000
#define NCP      1024
#define INV_TAU  10.0f

static __device__ __forceinline__ v8f wmma_bf16(v16bf a, v16bf b, v8f c) {
  return __builtin_amdgcn_wmma_f32_16x16x32_bf16(false, a, false, b, (short)0, c,
                                                 false, false);
}

// Load a 16x32 (MxK) bf16 A-operand from a row-major LDS tile (leading dim ld,
// starting K column koff), per CDNA5 16-bit A layout:
// lane m=lane&15, half=lane>>4; elems 0-7 hold K=koff+half*8+e, elems 8-15 hold
// K=koff+16+half*8+e.  (The B operand of Q*K^T has the identical per-lane layout
// when loaded from the key rows, so this loader serves both.)
static __device__ __forceinline__ v16bf load_a16x32(const __bf16* src, int row_base,
                                                    int ld, int koff) {
  int lane = threadIdx.x & 31;
  int m = lane & 15, half = lane >> 4;
  const __bf16* p = src + (row_base + m) * ld + koff + half * 8;
  v8bf lo = *(const v8bf*)p;
  v8bf hi = *(const v8bf*)(p + 16);
  return __builtin_shufflevector(lo, hi, 0, 1, 2, 3, 4, 5, 6, 7,
                                 8, 9, 10, 11, 12, 13, 14, 15);
}

// Load a 32x16 (KxN) bf16 B-operand from a row-major LDS tile [32][ld] starting
// at column col_base (B[k][n] = tile[k][col_base+n]): column gather.
static __device__ __forceinline__ v16bf load_b32x16(const __bf16* src, int col_base,
                                                    int ld) {
  int lane = threadIdx.x & 31;
  int n = lane & 15, half = lane >> 4;
  const __bf16* p = src + col_base + n;
  v16bf b;
#pragma unroll
  for (int e = 0; e < 8; ++e) b[e] = p[(half * 8 + e) * ld];
#pragma unroll
  for (int e = 0; e < 8; ++e) b[8 + e] = p[(16 + half * 8 + e) * ld];
  return b;
}

// ---------------- K1: row L2-normalize -> bf16 ----------------
__global__ void __launch_bounds__(128) k_normalize(const float* __restrict__ lbf,
                                                   const float* __restrict__ anchor,
                                                   __bf16* __restrict__ nbf) {
  int row = blockIdx.x;
  const float* src = (row < NLB) ? (lbf + (size_t)row * DIM)
                                 : (anchor + (size_t)(row - NLB) * DIM);
  __shared__ float red[128];
  int t = threadIdx.x;
  float v[4];
  float ss = 0.f;
#pragma unroll
  for (int i = 0; i < 4; ++i) { v[i] = src[t + i * 128]; ss += v[i] * v[i]; }
  red[t] = ss;
  __syncthreads();
  for (int s = 64; s > 0; s >>= 1) {
    if (t < s) red[t] += red[t + s];
    __syncthreads();
  }
  float inv = 1.0f / fmaxf(sqrtf(red[0]), 1e-12f);
#pragma unroll
  for (int i = 0; i < 4; ++i)
    nbf[(size_t)row * DIM + t + i * 128] = (__bf16)(v[i] * inv);
}

// ---------------- K2: class_val = [one_hot ; softmax(logits_u1)] -> bf16 ------
__global__ void __launch_bounds__(256) k_classval(const float* __restrict__ onehot,
                                                  const float* __restrict__ logits,
                                                  __bf16* __restrict__ vbf) {
  int row = blockIdx.x;
  int t = threadIdx.x;
  __shared__ float red[256];
  if (row < NLB) {
    for (int c = t; c < NCP; c += 256)
      vbf[(size_t)row * NCP + c] =
          (__bf16)((c < NC) ? onehot[(size_t)row * NC + c] : 0.f);
    return;
  }
  const float* src = logits + (size_t)(row - NLB) * NC;
  float v[4];
  float mx = -INFINITY;
#pragma unroll
  for (int i = 0; i < 4; ++i) {
    int c = t + i * 256;
    v[i] = (c < NC) ? src[c] : -INFINITY;
    mx = fmaxf(mx, v[i]);
  }
  red[t] = mx;
  __syncthreads();
  for (int s = 128; s > 0; s >>= 1) {
    if (t < s) red[t] = fmaxf(red[t], red[t + s]);
    __syncthreads();
  }
  mx = red[0];
  __syncthreads();
  float sum = 0.f;
#pragma unroll
  for (int i = 0; i < 4; ++i) {
    int c = t + i * 256;
    v[i] = (c < NC) ? __expf(v[i] - mx) : 0.f;
    sum += v[i];
  }
  red[t] = sum;
  __syncthreads();
  for (int s = 128; s > 0; s >>= 1) {
    if (t < s) red[t] += red[t + s];
    __syncthreads();
  }
  float inv = 1.0f / red[0];
#pragma unroll
  for (int i = 0; i < 4; ++i) {
    int c = t + i * 256;
    if (c < NCP) vbf[(size_t)row * NCP + c] = (__bf16)((c < NC) ? v[i] * inv : 0.f);
  }
}

// ---------------- K3: streaming row-softmax stats (m, l) over G = nn^T/tau ----
// Each WG owns 64 query rows; 8 waves tile a 64x64 S block as 4 rowgroups x
// 2 colgroups (each wave: 16 rows x 32 keys = two 16x16 WMMA accumulators).
__global__ void __launch_bounds__(256) k_stats(const __bf16* __restrict__ nbf,
                                               float* __restrict__ mrow_g,
                                               float* __restrict__ lrow_g) {
  __shared__ __bf16 sQ[64 * 64];
  __shared__ __bf16 sK[64 * 64];
  __shared__ float mbuf[2][64];
  __shared__ float lbuf[2][64];
  int rb = blockIdx.x;  // row block, 0..127
  int t = threadIdx.x;
  int lane = t & 31, w = t >> 5;
  int rg = w >> 1, cg = w & 1;
  int half = lane >> 4, nIdx = lane & 15;

  float m8[8], l8[8];
#pragma unroll
  for (int e = 0; e < 8; ++e) { m8[e] = -INFINITY; l8[e] = 0.f; }

  for (int kb = 0; kb < NROWS / 64; ++kb) {
    v8f acc0 = {};
    v8f acc1 = {};
    for (int dc = 0; dc < DIM / 64; ++dc) {
      const __bf16* qsrc = nbf + (size_t)rb * 64 * DIM + dc * 64;
      const __bf16* ksrc = nbf + (size_t)kb * 64 * DIM + dc * 64;
#pragma unroll
      for (int i = 0; i < 2; ++i) {  // 64x64 tile = 512 v8bf chunks
        int c = t + i * 256;
        int r = c >> 3, col = (c & 7) * 8;
        *(v8bf*)(sQ + r * 64 + col) = *(const v8bf*)(qsrc + (size_t)r * DIM + col);
        *(v8bf*)(sK + r * 64 + col) = *(const v8bf*)(ksrc + (size_t)r * DIM + col);
      }
      __syncthreads();
#pragma unroll
      for (int ks = 0; ks < 2; ++ks) {
        v16bf a = load_a16x32(sQ, rg * 16, 64, ks * 32);
        v16bf b0 = load_a16x32(sK, cg * 32, 64, ks * 32);
        v16bf b1 = load_a16x32(sK, cg * 32 + 16, 64, ks * 32);
        acc0 = wmma_bf16(a, b0, acc0);
        acc1 = wmma_bf16(a, b1, acc1);
      }
      __syncthreads();
    }
    // online logsumexp update (includes 1/tau scaling)
#pragma unroll
    for (int e = 0; e < 8; ++e) {
      float s0 = acc0[e] * INV_TAU, s1 = acc1[e] * INV_TAU;
      float nm = fmaxf(m8[e], fmaxf(s0, s1));
      l8[e] = l8[e] * __expf(m8[e] - nm) + __expf(s0 - nm) + __expf(s1 - nm);
      m8[e] = nm;
    }
  }
  // reduce across the 16 lanes of each half-group (same rows, different cols)
#pragma unroll
  for (int e = 0; e < 8; ++e) {
    float m = m8[e], l = l8[e];
    for (int off = 1; off < 16; off <<= 1) {
      float mo = __shfl_xor(m, off, 16);
      float lo = __shfl_xor(l, off, 16);
      float nm = fmaxf(m, mo);
      l = l * __expf(m - nm) + lo * __expf(mo - nm);
      m = nm;
    }
    m8[e] = m;
    l8[e] = l;
  }
  __syncthreads();
  if (nIdx == 0) {
#pragma unroll
    for (int e = 0; e < 8; ++e) {
      int row = rg * 16 + half * 8 + e;  // D-tile row = e + 8*half
      mbuf[cg][row] = m8[e];
      lbuf[cg][row] = l8[e];
    }
  }
  __syncthreads();
  if (t < 64) {
    float m0 = mbuf[0][t], m1 = mbuf[1][t];
    float nm = fmaxf(m0, m1);
    float l = lbuf[0][t] * __expf(m0 - nm) + lbuf[1][t] * __expf(m1 - nm);
    mrow_g[rb * 64 + t] = nm;
    lrow_g[rb * 64 + t] = l;
  }
}

// ---------------- K4: O = (exp(G - m)/l) @ class_val, flash-style -------------
// grid (8 col-blocks of 128, 128 row-blocks of 64). Per 32-key step: recompute
// S tile via WMMA, P=exp(...)->bf16 LDS, then P@V via WMMA (wave: 16x64 of O).
__global__ void __launch_bounds__(256) k_out(const __bf16* __restrict__ nbf,
                                             const __bf16* __restrict__ vbf,
                                             const float* __restrict__ mrow_g,
                                             const float* __restrict__ lrow_g,
                                             float* __restrict__ outO) {
  __shared__ __bf16 sQ[64 * 64];
  __shared__ __bf16 sK[32 * 64];
  __shared__ __bf16 sP[64 * 32];
  __shared__ __bf16 sV[32 * 128];
  __shared__ float sM[64];
  __shared__ float sLinv[64];
  int cb = blockIdx.x;  // 0..7 (128 output cols each)
  int rb = blockIdx.y;  // 0..127 (64 rows each)
  int t = threadIdx.x, lane = t & 31, w = t >> 5;
  int rg = w >> 1, cg = w & 1;
  int half = lane >> 4, nIdx = lane & 15;

  if (t < 64) {
    sM[t] = mrow_g[rb * 64 + t];
    sLinv[t] = 1.0f / lrow_g[rb * 64 + t];
  }
  __syncthreads();

  v8f o0 = {}, o1 = {}, o2 = {}, o3 = {};

  for (int kb = 0; kb < NROWS / 32; ++kb) {
    // ---- phase A: S tile (64 rows x 32 keys); wave = one 16x16 tile ----
    v8f sacc = {};
    for (int dc = 0; dc < DIM / 64; ++dc) {
      const __bf16* qsrc = nbf + (size_t)rb * 64 * DIM + dc * 64;
      const __bf16* ksrc = nbf + (size_t)kb * 32 * DIM + dc * 64;
#pragma unroll
      for (int i = 0; i < 2; ++i) {  // sQ: 512 chunks, 2/thread
        int c = t + i * 256;
        int r = c >> 3, col = (c & 7) * 8;
        *(v8bf*)(sQ + r * 64 + col) = *(const v8bf*)(qsrc + (size_t)r * DIM + col);
      }
      {  // sK: 256 chunks, 1/thread
        int r = t >> 3, col = (t & 7) * 8;
        *(v8bf*)(sK + r * 64 + col) = *(const v8bf*)(ksrc + (size_t)r * DIM + col);
      }
      __syncthreads();
#pragma unroll
      for (int ks = 0; ks < 2; ++ks) {
        v16bf a = load_a16x32(sQ, rg * 16, 64, ks * 32);
        v16bf b = load_a16x32(sK, cg * 16, 64, ks * 32);
        sacc = wmma_bf16(a, b, sacc);
      }
      __syncthreads();
    }
    // P = exp(s/tau - m) -> bf16 LDS (rows of this wave's 16x16 D tile)
#pragma unroll
    for (int e = 0; e < 8; ++e) {
      int r = rg * 16 + half * 8 + e;
      float p = __expf(sacc[e] * INV_TAU - sM[r]);
      sP[r * 32 + cg * 16 + nIdx] = (__bf16)p;
    }
    // V tile 32 x 128 -> LDS (512 chunks, 2/thread)
#pragma unroll
    for (int i = 0; i < 2; ++i) {
      int c = t + i * 256;
      int r = c >> 4, col = (c & 15) * 8;
      *(v8bf*)(sV + r * 128 + col) =
          *(const v8bf*)(vbf + (size_t)(kb * 32 + r) * NCP + cb * 128 + col);
    }
    __syncthreads();
    // ---- phase B: O += P @ V ; wave covers 16 rows x 64 cols ----
    {
      v16bf a = load_a16x32(sP, rg * 16, 32, 0);
      v16bf b;
      b = load_b32x16(sV, cg * 64 + 0, 128);  o0 = wmma_bf16(a, b, o0);
      b = load_b32x16(sV, cg * 64 + 16, 128); o1 = wmma_bf16(a, b, o1);
      b = load_b32x16(sV, cg * 64 + 32, 128); o2 = wmma_bf16(a, b, o2);
      b = load_b32x16(sV, cg * 64 + 48, 128); o3 = wmma_bf16(a, b, o3);
    }
    __syncthreads();
  }

  // scale by 1/l and store (rows map contiguously into d_out: lb then ulb1)
#pragma unroll
  for (int ti = 0; ti < 4; ++ti) {
    v8f ov = (ti == 0) ? o0 : (ti == 1) ? o1 : (ti == 2) ? o2 : o3;
#pragma unroll
    for (int e = 0; e < 8; ++e) {
      int rloc = rg * 16 + half * 8 + e;
      int row = rb * 64 + rloc;
      int col = cb * 128 + cg * 64 + ti * 16 + nIdx;
      if (col < NC) outO[(size_t)row * NC + col] = ov[e] * sLinv[rloc];
    }
  }
}

// ---------------- host side ----------------
extern "C" void kernel_launch(void* const* d_in, const int* in_sizes, int n_in,
                              void* d_out, int out_size, void* d_ws, size_t ws_size,
                              hipStream_t stream) {
  (void)in_sizes; (void)n_in; (void)out_size; (void)ws_size;
  const float* anchor   = (const float*)d_in[0];  // 6144x512
  const float* positive = (const float*)d_in[1];  // 6144x512
  const float* lbf      = (const float*)d_in[2];  // 2048x512
  const float* onehot   = (const float*)d_in[3];  // 2048x1000
  const float* logits_u1 = (const float*)d_in[5]; // 6144x1000
  const float* logits_u2 = (const float*)d_in[6]; // 6144x1000
  float* out = (float*)d_out;

  // workspace layout (~25.2 MB)
  __bf16* nbf  = (__bf16*)d_ws;                               // 8192x512 bf16
  __bf16* vbf  = (__bf16*)((char*)d_ws + 8388608);            // 8192x1024 bf16
  float*  mrow = (float*)((char*)d_ws + 25165824);            // 8192 f32
  float*  lrow = (float*)((char*)d_ws + 25198592);            // 8192 f32

  // output flat offsets (floats)
  const size_t O_ANCHOR = 0;
  const size_t O_POS    = 3145728;
  const size_t O_LBF    = 6291456;
  const size_t O_ONEHOT = 7340032;
  const size_t O_NEW    = 9388032;   // new_lb (2048x1000) + new_ulb1 (6144x1000)
  const size_t O_U2     = 17580032;

  hipMemcpyAsync(out + O_ANCHOR, anchor,   (size_t)3145728 * 4, hipMemcpyDeviceToDevice, stream);
  hipMemcpyAsync(out + O_POS,    positive, (size_t)3145728 * 4, hipMemcpyDeviceToDevice, stream);
  hipMemcpyAsync(out + O_LBF,    lbf,      (size_t)1048576 * 4, hipMemcpyDeviceToDevice, stream);
  hipMemcpyAsync(out + O_ONEHOT, onehot,   (size_t)2048000 * 4, hipMemcpyDeviceToDevice, stream);
  hipMemcpyAsync(out + O_U2,     logits_u2,(size_t)6144000 * 4, hipMemcpyDeviceToDevice, stream);

  k_normalize<<<NROWS, 128, 0, stream>>>(lbf, anchor, nbf);
  k_classval<<<NROWS, 256, 0, stream>>>(onehot, logits_u1, vbf);
  k_stats<<<NROWS / 64, 256, 0, stream>>>(nbf, mrow, lrow);
  k_out<<<dim3(8, NROWS / 64), 256, 0, stream>>>(nbf, vbf, mrow, lrow, out + O_NEW);
}
